// MixtureOfExpertsLayer_21251498181443
// MI455X (gfx1250) — matrix operand order repaired and checked by reading, not compiled, not run on hardware
//
#include <hip/hip_runtime.h>
#include <hip/hip_bf16.h>
#include <cstdint>

typedef __bf16 bf16;
typedef __attribute__((ext_vector_type(16))) __bf16 bf16x16;
typedef __attribute__((ext_vector_type(8)))  float  f32x8;
typedef __attribute__((ext_vector_type(4)))  unsigned int v4u;
typedef __attribute__((ext_vector_type(8)))  int          v8i;
typedef __attribute__((ext_vector_type(4)))  int          v4i;

// Problem sizes from the reference: B=2, S=2048 -> T=4096; D=1024, F=4096, E=8, top-2.
constexpr int T  = 4096;
constexpr int D  = 1024;
constexpr int F  = 4096;
constexpr int E  = 8;
constexpr int MT = 16;    // tokens per M-tile
constexpr int FC = 1024;  // F processed per chunk
constexpr int KB = 64;    // K rows staged per TDM tile

// ---- CDNA5 Tensor Data Mover availability (6-arg clang-23 builtin) ----
#if defined(__has_builtin)
#if __has_builtin(__builtin_amdgcn_tensor_load_to_lds) && \
    __has_builtin(__builtin_amdgcn_s_wait_tensorcnt)
#define MOE_TDM 1
#endif
#endif
#ifndef MOE_TDM
#define MOE_TDM 0
#endif

__device__ __forceinline__ float fast_silu(float v) {
  // v * sigmoid(v) with v_rcp_f32 instead of IEEE divide
  return v * __builtin_amdgcn_rcpf(1.f + __expf(-v));
}

// ---------------- fragment loaders (CDNA5 wave32 WMMA layouts) ----------------

// A-matrix 16x32 bf16 (MxK): lane&15 = row M; lane>>4 selects K-half.
__device__ __forceinline__ bf16x16 load_a_frag_lds(const bf16* __restrict__ src,
                                                   int ld, int k0, int lane) {
  bf16x16 a;
  const int m    = lane & 15;
  const int half = lane >> 4;
  const bf16* row = src + m * ld + k0;
#pragma unroll
  for (int e = 0; e < 16; ++e) {
    const int k = half * 16 + e;
    a[e] = row[k];
  }
  return a;
}

// B-matrix 32x16 bf16 (KxN) from a row-major fp32 global matrix (leading dim ld).
__device__ __forceinline__ bf16x16 load_b_frag_g(const float* __restrict__ src,
                                                 size_t ld, int k0, int n0, int lane) {
  bf16x16 b;
  const int n     = lane & 15;
  const int khalf = lane >> 4;
  const float* p  = src + (size_t)(k0 + khalf * 16) * ld + (size_t)(n0 + n);
#pragma unroll
  for (int e = 0; e < 16; ++e) b[e] = (bf16)p[(size_t)e * ld];
  return b;
}

#if MOE_TDM
// B-matrix 32x16 fragment from a per-wave fp32 LDS staging tile [KB][16].
__device__ __forceinline__ bf16x16 load_b_frag_wb(const float* __restrict__ wbw,
                                                  int k0, int lane) {
  bf16x16 b;
  const int n     = lane & 15;
  const int khalf = lane >> 4;
  const float* p  = wbw + (k0 + khalf * 16) * 16 + n;
#pragma unroll
  for (int e = 0; e < 16; ++e) b[e] = (bf16)p[e * 16];
  return b;
}

// Issue one TDM load of a [rows x cols] fp32 tile (row stride = ld elements)
// from global into LDS at byte address lds_addr. 2D descriptor, groups 2/3 zero.
__device__ __forceinline__ void tdm_load_tile(const float* gsrc, unsigned lds_addr,
                                              unsigned ld, unsigned rows, unsigned cols) {
  const unsigned long long ga = (unsigned long long)(size_t)gsrc;
  v4u g0;
  g0[0] = 1u;                                            // count=1, user descriptor
  g0[1] = lds_addr;                                      // lds_addr (bytes)
  g0[2] = (unsigned)ga;                                  // global_addr[31:0]
  g0[3] = (unsigned)((ga >> 32) & 0x1FFFFFFull)          // global_addr[56:32]
        | (2u << 30);                                    // type=2 ("image")
  const unsigned td0 = ld;                               // tensor_dim0 (elements)
  const unsigned td1 = 1u << 20;                         // tensor_dim1: large (no OOB clip)
  const unsigned long long s0 = (unsigned long long)ld;  // tensor_dim0_stride
  v8i g1;
  g1[0] = (int)(2u << 16);                               // workgroup_mask=0, data_size=4B
  g1[1] = (int)((td0 & 0xFFFFu) << 16);                  // abar=0 | tensor_dim0[15:0]
  g1[2] = (int)((td0 >> 16) | ((td1 & 0xFFFFu) << 16));  // td0[31:16] | td1[15:0]
  g1[3] = (int)((td1 >> 16) | (cols << 16));             // td1[31:16] | tile_dim0
  g1[4] = (int)rows;                                     // tile_dim1 | tile_dim2=0
  g1[5] = (int)(unsigned)s0;                             // stride0[31:0]
  g1[6] = (int)((unsigned)(s0 >> 32) & 0xFFFFu);         // stride0[47:32] | stride1[15:0]=0
  g1[7] = 0;                                             // stride1[47:16]=0
  const v4i z4 = {0, 0, 0, 0};
  const v8i z8 = {0, 0, 0, 0, 0, 0, 0, 0};
  __builtin_amdgcn_tensor_load_to_lds(g0, g1, z4, z4, z8, 0);
}

__device__ __forceinline__ void wait_lds_reads_done() {
  asm volatile("s_wait_dscnt 0" ::: "memory");
}

// One KB-deep WMMA sweep: A from LDS (bf16), B from a staged fp32 LDS tile.
__device__ __forceinline__ void mm_step(const bf16* __restrict__ asrc, int ald, int kb,
                                        const float* __restrict__ wbw, int lane,
                                        f32x8& acc) {
#pragma unroll
  for (int kk = 0; kk < KB; kk += 32) {
    bf16x16 a = load_a_frag_lds(asrc, ald, kb + kk, lane);
    bf16x16 b = load_b_frag_wb(wbw, kk, lane);
    acc = __builtin_amdgcn_wmma_f32_16x16x32_bf16(false, a, false, b,
                                                  (short)0, acc, false, false);
  }
}
#endif // MOE_TDM

// ---------------- kernel 1: zero output + expert counters ----------------

__global__ void moe_zero(float* __restrict__ out, int n4, int* __restrict__ cnt) {
  int i = blockIdx.x * blockDim.x + threadIdx.x;
  if (i < n4) reinterpret_cast<float4*>(out)[i] = make_float4(0.f, 0.f, 0.f, 0.f);
  if (i < E) cnt[i] = 0;
}

// ---------------- kernel 2: router (one wave32 per token) ----------------

__global__ void moe_router(const float* __restrict__ x, const float* __restrict__ Wr,
                           int* __restrict__ cnt, int* __restrict__ tlist,
                           float* __restrict__ twt) {
  const int wave = threadIdx.x >> 5;
  const int lane = threadIdx.x & 31;
  const int t = blockIdx.x * 8 + wave;
  if (t >= T) return;

  float acc[E];
#pragma unroll
  for (int e = 0; e < E; ++e) acc[e] = 0.f;

  const float* xr = x + (size_t)t * D;
  for (int d = lane; d < D; d += 32) {
    const float xv = xr[d];
#pragma unroll
    for (int e = 0; e < E; ++e) acc[e] += xv * Wr[e * D + d];
  }
#pragma unroll
  for (int e = 0; e < E; ++e) {
#pragma unroll
    for (int m = 16; m >= 1; m >>= 1) acc[e] += __shfl_xor(acc[e], m, 32);
  }

  if (lane == 0) {
    int i1 = 0;
#pragma unroll
    for (int e = 1; e < E; ++e) if (acc[e] > acc[i1]) i1 = e;
    int i2 = (i1 == 0) ? 1 : 0;
#pragma unroll
    for (int e = 0; e < E; ++e) if (e != i1 && acc[e] > acc[i2]) i2 = e;
    // softmax over 8 then renormalize top-2 == 2-way softmax of the top-2 logits
    const float w1 = __builtin_amdgcn_rcpf(1.f + __expf(acc[i2] - acc[i1]));
    const float w2 = 1.f - w1;
    int p1 = atomicAdd(&cnt[i1], 1);
    tlist[i1 * T + p1] = t; twt[i1 * T + p1] = w1;
    int p2 = atomicAdd(&cnt[i2], 1);
    tlist[i2 * T + p2] = t; twt[i2 * T + p2] = w2;
  }
}

// ---------------- kernel 3: grouped FFN (fused SiLU(X W1) W2) ----------------

__global__ __launch_bounds__(256) void moe_ffn(const float* __restrict__ x,
                                               const float* __restrict__ W1,
                                               const float* __restrict__ W2,
                                               const int* __restrict__ cnt,
                                               const int* __restrict__ tlist,
                                               const float* __restrict__ twt,
                                               float* __restrict__ out) {
  __shared__ bf16  xs[MT * D];          // 32 KB: gathered X tile (bf16)
  __shared__ bf16  hs[MT * FC];         // 32 KB: SiLU(X W1) chunk (bf16)
#if MOE_TDM
  __shared__ float wb[8][2][KB * 16];   // 64 KB: per-wave double-buffered TDM staging
#endif
  __shared__ int   ts[MT];
  __shared__ float ws[MT];

  const int e    = blockIdx.y;
  const int tile = blockIdx.x;
  const int n    = cnt[e];
  if (tile * MT >= n) return;

  const int tid  = threadIdx.x;
  const int wave = tid >> 5;
  const int lane = tid & 31;

  if (tid < MT) {
    const int g = tile * MT + tid;
    if (g < n) { ts[tid] = tlist[e * T + g]; ws[tid] = twt[e * T + g]; }
    else       { ts[tid] = -1;               ws[tid] = 0.f; }
  }
  __syncthreads();

  // gather X tile -> LDS bf16 (coalesced: 256 threads x 64 elements)
  for (int i = tid; i < MT * D; i += 256) {
    const int r = i >> 10;          // i / D
    const int c = i & (D - 1);      // i % D
    const int t = ts[r];
    xs[i] = (t >= 0) ? (bf16)x[(size_t)t * D + c] : (bf16)0.f;
  }
  __syncthreads();

  const float* w1e = W1 + (size_t)e * D * F;   // [D, F]
  const float* w2e = W2 + (size_t)e * F * D;   // [F, D]

#if MOE_TDM
  // Static double-buffer pointers (no runtime indexing -> stays in LDS addressing)
  const float* wbw0   = &wb[wave][0][0];
  const float* wbw1   = &wb[wave][1][0];
  const unsigned wba0 = (unsigned)(size_t)(const void*)wbw0;
  const unsigned wba1 = (unsigned)(size_t)(const void*)wbw1;
#endif

  // persistent Y accumulators: wave covers 128 cols of D -> 8 tiles of 16x16 f32
  f32x8 yacc[8];
#pragma unroll
  for (int j = 0; j < 8; ++j)
#pragma unroll
    for (int r = 0; r < 8; ++r) yacc[j][r] = 0.f;

  const int colLane = lane & 15;
  const int rowBase = (lane >> 4) * 8;

  for (int fc = 0; fc < F; fc += FC) {
    // ---- phase 1: H = SiLU(X @ W1[:, fc:fc+FC]) ----
    for (int j = 0; j < 8; ++j) {
      const int ncol = wave * 128 + j * 16;   // column inside chunk
      f32x8 acc;
#pragma unroll
      for (int r = 0; r < 8; ++r) acc[r] = 0.f;
#if MOE_TDM
      wait_lds_reads_done();
      tdm_load_tile(w1e + (fc + ncol), wba0, F, KB, 16);    // prologue -> buf0
      for (int kb = 0; kb < D; kb += 2 * KB) {
        // prefetch kb+KB -> buf1, then compute buf0
        wait_lds_reads_done();
        tdm_load_tile(w1e + (size_t)(kb + KB) * F + (fc + ncol), wba1, F, KB, 16);
        __builtin_amdgcn_s_wait_tensorcnt((short)1);
        mm_step(xs, D, kb, wbw0, lane, acc);
        // prefetch kb+2KB -> buf0 (unless last), then compute buf1
        if (kb + 2 * KB < D) {
          wait_lds_reads_done();
          tdm_load_tile(w1e + (size_t)(kb + 2 * KB) * F + (fc + ncol), wba0, F, KB, 16);
          __builtin_amdgcn_s_wait_tensorcnt((short)1);
        } else {
          __builtin_amdgcn_s_wait_tensorcnt((short)0);
        }
        mm_step(xs, D, kb + KB, wbw1, lane, acc);
      }
#else
      for (int kb = 0; kb < D; kb += 32) {
        bf16x16 a = load_a_frag_lds(xs, D, kb, lane);
        bf16x16 b = load_b_frag_g(w1e, (size_t)F, kb, fc + ncol, lane);
        acc = __builtin_amdgcn_wmma_f32_16x16x32_bf16(false, a, false, b,
                                                      (short)0, acc, false, false);
      }
#endif
      const int col = ncol + colLane;
#pragma unroll
      for (int r = 0; r < 8; ++r)
        hs[(rowBase + r) * FC + col] = (bf16)fast_silu(acc[r]);
    }
    __syncthreads();

    // ---- phase 2: Y += H_chunk @ W2[fc:fc+FC, :] ----
    for (int j = 0; j < 8; ++j) {
      const int dcol0 = wave * 128 + j * 16;
#if MOE_TDM
      wait_lds_reads_done();
      tdm_load_tile(w2e + (size_t)fc * D + dcol0, wba0, D, KB, 16);   // prologue -> buf0
      for (int kb = 0; kb < FC; kb += 2 * KB) {
        wait_lds_reads_done();
        tdm_load_tile(w2e + (size_t)(fc + kb + KB) * D + dcol0, wba1, D, KB, 16);
        __builtin_amdgcn_s_wait_tensorcnt((short)1);
        mm_step(hs, FC, kb, wbw0, lane, yacc[j]);
        if (kb + 2 * KB < FC) {
          wait_lds_reads_done();
          tdm_load_tile(w2e + (size_t)(fc + kb + 2 * KB) * D + dcol0, wba0, D, KB, 16);
          __builtin_amdgcn_s_wait_tensorcnt((short)1);
        } else {
          __builtin_amdgcn_s_wait_tensorcnt((short)0);
        }
        mm_step(hs, FC, kb + KB, wbw1, lane, yacc[j]);
      }
#else
      for (int kb = 0; kb < FC; kb += 32) {
        bf16x16 a = load_a_frag_lds(hs, FC, kb, lane);
        bf16x16 b = load_b_frag_g(w2e, (size_t)D, fc + kb, dcol0, lane);
        yacc[j] = __builtin_amdgcn_wmma_f32_16x16x32_bf16(false, a, false, b,
                                                          (short)0, yacc[j], false, false);
      }
#endif
    }
    __syncthreads();   // hs reused next chunk
  }

  // ---- scatter-accumulate weighted outputs ----
#pragma unroll
  for (int j = 0; j < 8; ++j) {
    const int dcol = wave * 128 + j * 16 + colLane;
#pragma unroll
    for (int r = 0; r < 8; ++r) {
      const int row = rowBase + r;
      const int t = ts[row];
      if (t >= 0)
        atomicAdd(&out[(size_t)t * D + dcol], ws[row] * yacc[j][r]);
    }
  }
}

// ---------------- host launcher ----------------

extern "C" void kernel_launch(void* const* d_in, const int* in_sizes, int n_in,
                              void* d_out, int out_size, void* d_ws, size_t ws_size,
                              hipStream_t stream) {
  const float* x  = (const float*)d_in[0];   // [B,S,D]
  const float* Wr = (const float*)d_in[1];   // [E,D]
  const float* W1 = (const float*)d_in[2];   // [E,D,F]
  const float* W2 = (const float*)d_in[3];   // [E,F,D]
  float* out = (float*)d_out;

  char* ws = (char*)d_ws;
  int*   cnt   = (int*)ws;                                    // [E]
  int*   tlist = (int*)(ws + 256);                            // [E*T]
  float* twt   = (float*)(ws + 256 + sizeof(int) * E * T);    // [E*T]

  const int n4 = (T * D) / 4;
  moe_zero<<<(n4 + 255) / 256, 256, 0, stream>>>(out, n4, cnt);
  moe_router<<<(T + 7) / 8, 256, 0, stream>>>(x, Wr, cnt, tlist, twt);
  dim3 grid(T / MT, E);
  moe_ffn<<<grid, 256, 0, stream>>>(x, W1, W2, cnt, tlist, twt, out);
}
